// CausalSelfAttention_34660386079268
// MI455X (gfx1250) — compile-verified, matrix operand order checked by reading, not run on hardware
//
#include <hip/hip_runtime.h>
#include <cstddef>

// ---------------------------------------------------------------------------
// CausalSelfAttention for MI455X (gfx1250): bf16 WMMA GEMMs + flash attention
// B=2, T=2048, C=1024, H=16, HS=64
// ---------------------------------------------------------------------------

#define B_  2
#define T_  2048
#define C_  1024
#define H_  16
#define HS_ 64
#define M_  (B_ * T_)          // 4096
#define N_QKV (3 * C_)         // 3072

typedef __attribute__((ext_vector_type(16))) __bf16 v16bf;
typedef __attribute__((ext_vector_type(8)))  float  v8f;
typedef __attribute__((ext_vector_type(4)))  int    v4i;

__device__ __forceinline__ unsigned short f32_to_bf16_raw(float f) {
    unsigned int u = __float_as_uint(f);
    unsigned int lsb = (u >> 16) & 1u;
    u += 0x7fffu + lsb;                 // round-to-nearest-even
    return (unsigned short)(u >> 16);
}

__device__ __forceinline__ v16bf load_frag(const unsigned short* p0,
                                           const unsigned short* p1) {
    v16bf r;
    uint4* h = reinterpret_cast<uint4*>(&r);
    h[0] = *reinterpret_cast<const uint4*>(p0);
    h[1] = *reinterpret_cast<const uint4*>(p1);
    return r;
}

__device__ __forceinline__ v8f wmma_bf16(v16bf a, v16bf b, v8f c) {
    return __builtin_amdgcn_wmma_f32_16x16x32_bf16(
        /*neg_a=*/false, a, /*neg_b=*/false, b,
        /*c_mod=*/(short)0, c, /*reuse_a=*/false, /*reuse_b=*/false);
}

// ---------------------------------------------------------------------------
// Async global->LDS copy (CDNA5 GLOBAL_LOAD_ASYNC_TO_LDS_B128, ASYNCcnt).
// Probe-derived signature: (v4i AS1* gsrc, v4i AS3* ldst, imm offset, imm cpol)
// Falls back to a register round-trip if the builtin is absent.
// ---------------------------------------------------------------------------
#if __has_builtin(__builtin_amdgcn_global_load_async_to_lds_b128)
#define HAVE_ASYNC_LDS 1
typedef __attribute__((address_space(1))) v4i as1_v4i;
typedef __attribute__((address_space(3))) v4i as3_v4i;
#else
#define HAVE_ASYNC_LDS 0
#endif

__device__ __forceinline__ void copy16_g2lds(const unsigned short* g,
                                             unsigned short* l) {
#if HAVE_ASYNC_LDS
    __builtin_amdgcn_global_load_async_to_lds_b128(
        (as1_v4i*)g, (as3_v4i*)l, 0, 0);
#else
    *reinterpret_cast<uint4*>(l) = *reinterpret_cast<const uint4*>(g);
#endif
}

__device__ __forceinline__ void async_lds_wait() {
#if HAVE_ASYNC_LDS
#if __has_builtin(__builtin_amdgcn_s_wait_asynccnt)
    __builtin_amdgcn_s_wait_asynccnt(0);
#else
    asm volatile("s_wait_asynccnt 0" ::: "memory");
#endif
#endif
}

// ---------------------------------------------------------------------------
// fp32 -> bf16 cast
// ---------------------------------------------------------------------------
__global__ __launch_bounds__(256) void cast_f32_bf16_kernel(
    const float* __restrict__ src, unsigned short* __restrict__ dst, int n) {
    int i = blockIdx.x * 256 + threadIdx.x;
    if (i < n) dst[i] = f32_to_bf16_raw(src[i]);
}

// ---------------------------------------------------------------------------
// Shared GEMM mainloop: 128x128 block tile, K-step 32, 8 waves (4x2 grid).
// A: [M x K] bf16 row-major (staged via async LDS copy).
// Bw: [K x N] bf16 row-major (staged transposed through registers).
// Each wave owns a 32x64 tile = acc[2][4] of 16x16 f32.
// ---------------------------------------------------------------------------
#define LDA 40
#define LDB 40

__device__ __forceinline__ void gemm_mainloop(
    const unsigned short* __restrict__ A,
    const unsigned short* __restrict__ Bw,
    int Ktot, int Ntot, int m0, int n0,
    unsigned short* ldsA, unsigned short* ldsBt,
    v8f acc[2][4]) {

    const int tid   = threadIdx.x;        // 0..255
    const int lane  = tid & 31;
    const int wv    = tid >> 5;           // 0..7
    const int waveM = wv >> 1;            // 0..3
    const int waveN = wv & 1;             // 0..1
    const int row16 = lane & 15;
    const int kg    = lane >> 4;          // 0/1 (K-group of the wave half)

    const int arow = tid >> 1;            // 0..127
    const int acol = (tid & 1) << 4;      // 0 / 16
    const int bkr  = tid >> 3;            // 0..31
    const int bnc  = (tid & 7) << 4;      // 0..112

    for (int k0 = 0; k0 < Ktot; k0 += 32) {
        // ---- B tile into registers (needs transpose, cannot go async) --------
        const uint4* gB = reinterpret_cast<const uint4*>(
            Bw + (size_t)(k0 + bkr) * Ntot + n0 + bnc);
        uint4 b0 = gB[0], b1 = gB[1];

        if (k0 + 32 < Ktot) {  // speculative prefetch of next K-slab
            __builtin_prefetch(A + (size_t)(m0 + arow) * Ktot + k0 + 32 + acol);
            __builtin_prefetch(Bw + (size_t)(k0 + 32 + bkr) * Ntot + n0 + bnc);
        }

        __syncthreads();   // previous compute done before overwriting LDS

        // ---- A tile: async global->LDS (row-major [128][LDA]) ----------------
        const unsigned short* gArow = A + (size_t)(m0 + arow) * Ktot + k0 + acol;
        copy16_g2lds(gArow,     &ldsA[arow * LDA + acol]);
        copy16_g2lds(gArow + 8, &ldsA[arow * LDA + acol + 8]);

        // ---- B tile: transposed [128 n][LDB k] so frag loads are contiguous --
        unsigned short tmp[16];
        *reinterpret_cast<uint4*>(tmp)     = b0;
        *reinterpret_cast<uint4*>(tmp + 8) = b1;
#pragma unroll
        for (int j = 0; j < 16; ++j)
            ldsBt[(bnc + j) * LDB + bkr] = tmp[j];

        async_lds_wait();
        __syncthreads();

        // ---- compute: 2x4 WMMA tiles per wave --------------------------------
        v16bf af[2];
#pragma unroll
        for (int mi = 0; mi < 2; ++mi) {
            const unsigned short* pa =
                &ldsA[(waveM * 32 + mi * 16 + row16) * LDA + kg * 8];
            af[mi] = load_frag(pa, pa + 16);
        }
#pragma unroll
        for (int ni = 0; ni < 4; ++ni) {
            const unsigned short* pb =
                &ldsBt[(waveN * 64 + ni * 16 + row16) * LDB + kg * 8];
            v16bf bfrag = load_frag(pb, pb + 16);
#pragma unroll
            for (int mi = 0; mi < 2; ++mi)
                acc[mi][ni] = wmma_bf16(af[mi], bfrag, acc[mi][ni]);
        }
    }
}

// ---------------------------------------------------------------------------
// QKV GEMM: x_bf16[4096,1024] @ Wattn_bf16[1024,3072] + b_attn
// Epilogue scatters bf16 q/k/v into head-major [B,H,T,HS].
// grid: 32 (M tiles) * 24 (N tiles) = 768 blocks x 256 threads
// ---------------------------------------------------------------------------
__global__ __launch_bounds__(256) void qkv_gemm_kernel(
    const unsigned short* __restrict__ A,
    const unsigned short* __restrict__ Bw,
    const float* __restrict__ bias,
    unsigned short* __restrict__ qo,
    unsigned short* __restrict__ ko,
    unsigned short* __restrict__ vo) {

    __shared__ unsigned short ldsA[128 * LDA];
    __shared__ unsigned short ldsBt[128 * LDB];

    const int bid = blockIdx.x;
    const int m0 = (bid / 24) * 128;
    const int n0 = (bid % 24) * 128;

    v8f zero = {};
    v8f acc[2][4];
#pragma unroll
    for (int mi = 0; mi < 2; ++mi)
#pragma unroll
        for (int ni = 0; ni < 4; ++ni) acc[mi][ni] = zero;

    gemm_mainloop(A, Bw, C_, N_QKV, m0, n0, ldsA, ldsBt, acc);

    const int lane  = threadIdx.x & 31;
    const int wv    = threadIdx.x >> 5;
    const int waveM = wv >> 1;
    const int waveN = wv & 1;
    const int row16 = lane & 15;
    const int hi    = lane >> 4;

#pragma unroll
    for (int mi = 0; mi < 2; ++mi) {
#pragma unroll
        for (int ni = 0; ni < 4; ++ni) {
            int gn = n0 + waveN * 64 + ni * 16 + row16;
            float bv = bias[gn];
            int which = gn >> 10;          // 0=q 1=k 2=v
            int c = gn & (C_ - 1);
            int h = c >> 6;
            int d = c & (HS_ - 1);
            unsigned short* dst = (which == 0) ? qo : (which == 1) ? ko : vo;
            int gm_base = m0 + waveM * 32 + mi * 16 + (hi << 3);
#pragma unroll
            for (int r = 0; r < 8; ++r) {
                int gm = gm_base + r;
                int b = gm >> 11;          // / T_
                int t = gm & (T_ - 1);
                float val = acc[mi][ni][r] + bv;
                dst[(((size_t)(b * H_ + h)) * T_ + t) * HS_ + d] =
                    f32_to_bf16_raw(val);
            }
        }
    }
}

// ---------------------------------------------------------------------------
// Flash attention: one block = 4 waves = 64 q rows of one (b,h).
// K/V streamed in 32-row tiles through LDS; online softmax; never
// materializes the TxT score matrix.
// grid: B*H*(T/64) = 1024 blocks x 128 threads
// ---------------------------------------------------------------------------
#define VLD 40

__global__ __launch_bounds__(128) void flash_attn_kernel(
    const unsigned short* __restrict__ q,
    const unsigned short* __restrict__ k,
    const unsigned short* __restrict__ v,
    unsigned short* __restrict__ y) {

    __shared__ unsigned short ldsK[32 * 64];     // [k-row][d] row-major
    __shared__ unsigned short ldsVt[64 * VLD];   // [d][k-row] transposed
    __shared__ unsigned short ldsP[4 * 16 * 32]; // per-wave P scratch

    const int tid   = threadIdx.x;     // 0..127
    const int lane  = tid & 31;
    const int wv    = tid >> 5;        // 0..3
    const int row16 = lane & 15;
    const int hi    = lane >> 4;       // 0/1
    const int kg    = hi;

    const int bh = blockIdx.x >> 5;            // 0..31 (B*H)
    const int qt = (blockIdx.x & 31) << 6;     // q-tile base (64 rows)
    const int q0 = qt + wv * 16;               // this wave's q rows
    const size_t headbase = (size_t)bh * T_ * HS_;

    // Q fragments (kept in registers the whole kernel)
    v16bf aQ[2];
    {
        const unsigned short* pq =
            q + headbase + (size_t)(q0 + row16) * HS_ + kg * 8;
        aQ[0] = load_frag(pq,      pq + 16);   // d = 0..31
        aQ[1] = load_frag(pq + 32, pq + 48);   // d = 32..63
    }

    v8f zero = {};
    v8f o[4] = {zero, zero, zero, zero};
    float mrow[8], lrow[8];
#pragma unroll
    for (int r = 0; r < 8; ++r) { mrow[r] = -1e30f; lrow[r] = 0.0f; }

    const int kend = qt + 64;                  // causal bound for this block
    const int skr = tid >> 2;                  // 0..31 (staging k-row)
    const int sdc = (tid & 3) << 4;            // 0,16,32,48 (staging d-col)
    const float scale = 0.125f;                // 1/sqrt(64)

    for (int k0 = 0; k0 < kend; k0 += 32) {
        // ---- V tile into registers (transposed store, cannot go async) -------
        const uint4* gv = reinterpret_cast<const uint4*>(
            v + headbase + (size_t)(k0 + skr) * HS_ + sdc);
        uint4 vv0 = gv[0], vv1 = gv[1];

        __syncthreads();

        // ---- K tile: async global->LDS, row-major [32][64] -------------------
        const unsigned short* gk =
            k + headbase + (size_t)(k0 + skr) * HS_ + sdc;
        copy16_g2lds(gk,     &ldsK[skr * 64 + sdc]);
        copy16_g2lds(gk + 8, &ldsK[skr * 64 + sdc + 8]);

        // ---- V tile: transposed [64 d][VLD k] --------------------------------
        unsigned short tv[16];
        *reinterpret_cast<uint4*>(tv)     = vv0;
        *reinterpret_cast<uint4*>(tv + 8) = vv1;
#pragma unroll
        for (int j = 0; j < 16; ++j)
            ldsVt[(sdc + j) * VLD + skr] = tv[j];

        async_lds_wait();
        __syncthreads();

        // waves fully above the causal frontier skip compute (no barriers here)
        if (k0 < q0 + 16) {
            // ---- S = Q K^T : 16(q) x 32(k) as two 16x16 tiles ----------------
            v8f s[2];
#pragma unroll
            for (int nh = 0; nh < 2; ++nh) {
                const unsigned short* pk =
                    &ldsK[(nh * 16 + row16) * 64 + kg * 8];
                v16bf bK0 = load_frag(pk,      pk + 16);   // d 0..31
                v16bf bK1 = load_frag(pk + 32, pk + 48);   // d 32..63
                v8f sv = zero;
                sv = wmma_bf16(aQ[0], bK0, sv);
                sv = wmma_bf16(aQ[1], bK1, sv);
                s[nh] = sv;
            }

            // ---- causal mask + online softmax update -------------------------
            float p0[8], p1[8], corr[8];
#pragma unroll
            for (int r = 0; r < 8; ++r) {
                int qg  = q0 + r + (hi << 3);
                int kg0 = k0 + row16;
                int kg1 = k0 + 16 + row16;
                float s0 = s[0][r] * scale; if (kg0 > qg) s0 = -1e30f;
                float s1 = s[1][r] * scale; if (kg1 > qg) s1 = -1e30f;
                float rm = fmaxf(s0, s1);
                rm = fmaxf(rm, __shfl_xor(rm, 1));
                rm = fmaxf(rm, __shfl_xor(rm, 2));
                rm = fmaxf(rm, __shfl_xor(rm, 4));
                rm = fmaxf(rm, __shfl_xor(rm, 8));
                float mnew = fmaxf(mrow[r], rm);
                float cr = __expf(mrow[r] - mnew);
                mrow[r] = mnew;
                float e0 = __expf(s0 - mnew);
                float e1 = __expf(s1 - mnew);
                float rs = e0 + e1;
                rs += __shfl_xor(rs, 1);
                rs += __shfl_xor(rs, 2);
                rs += __shfl_xor(rs, 4);
                rs += __shfl_xor(rs, 8);
                lrow[r] = lrow[r] * cr + rs;
                corr[r] = cr;
                p0[r] = e0;
                p1[r] = e1;
            }
#pragma unroll
            for (int nd = 0; nd < 4; ++nd)
#pragma unroll
                for (int r = 0; r < 8; ++r) o[nd][r] *= corr[r];

            // ---- redistribute P (D-layout -> A-layout) via LDS ---------------
            unsigned short* pw = &ldsP[wv * 512];
#pragma unroll
            for (int r = 0; r < 8; ++r) {
                int rowM = r + (hi << 3);
                pw[rowM * 32 + row16]      = f32_to_bf16_raw(p0[r]);
                pw[rowM * 32 + 16 + row16] = f32_to_bf16_raw(p1[r]);
            }
            asm volatile("s_wait_dscnt 0" ::: "memory");  // wave-local LDS RAW
            const unsigned short* pr = &ldsP[wv * 512 + row16 * 32 + kg * 8];
            v16bf aP = load_frag(pr, pr + 16);

            // ---- O += P V : 4 WMMAs over the 64-wide d dimension -------------
#pragma unroll
            for (int nd = 0; nd < 4; ++nd) {
                const unsigned short* pv =
                    &ldsVt[(nd * 16 + row16) * VLD + kg * 8];
                v16bf bV = load_frag(pv, pv + 16);
                o[nd] = wmma_bf16(aP, bV, o[nd]);
            }
        }
    }

    // ---- normalize and store y as bf16 in [B,T,C] layout ---------------------
    const int b = bh >> 4;
    const int h = bh & 15;
#pragma unroll
    for (int r = 0; r < 8; ++r) {
        float inv = 1.0f / lrow[r];
        int t = q0 + r + (hi << 3);
        size_t rowoff = ((size_t)(b * T_ + t)) * C_ + h * HS_;
#pragma unroll
        for (int nd = 0; nd < 4; ++nd)
            y[rowoff + nd * 16 + row16] = f32_to_bf16_raw(o[nd][r] * inv);
    }
}

// ---------------------------------------------------------------------------
// Output projection: y_bf16[4096,1024] @ Wproj_bf16[1024,1024] + b_proj -> f32
// grid: 32 * 8 = 256 blocks x 256 threads
// ---------------------------------------------------------------------------
__global__ __launch_bounds__(256) void proj_gemm_kernel(
    const unsigned short* __restrict__ A,
    const unsigned short* __restrict__ Bw,
    const float* __restrict__ bias,
    float* __restrict__ out) {

    __shared__ unsigned short ldsA[128 * LDA];
    __shared__ unsigned short ldsBt[128 * LDB];

    const int bid = blockIdx.x;
    const int m0 = (bid / 8) * 128;
    const int n0 = (bid % 8) * 128;

    v8f zero = {};
    v8f acc[2][4];
#pragma unroll
    for (int mi = 0; mi < 2; ++mi)
#pragma unroll
        for (int ni = 0; ni < 4; ++ni) acc[mi][ni] = zero;

    gemm_mainloop(A, Bw, C_, C_, m0, n0, ldsA, ldsBt, acc);

    const int lane  = threadIdx.x & 31;
    const int wv    = threadIdx.x >> 5;
    const int waveM = wv >> 1;
    const int waveN = wv & 1;
    const int row16 = lane & 15;
    const int hi    = lane >> 4;

#pragma unroll
    for (int mi = 0; mi < 2; ++mi) {
#pragma unroll
        for (int ni = 0; ni < 4; ++ni) {
            int gn = n0 + waveN * 64 + ni * 16 + row16;
            float bv = bias[gn];
            int gm_base = m0 + waveM * 32 + mi * 16 + (hi << 3);
#pragma unroll
            for (int r = 0; r < 8; ++r) {
                int gm = gm_base + r;
                out[(size_t)gm * C_ + gn] = acc[mi][ni][r] + bv;
            }
        }
    }
}

// ---------------------------------------------------------------------------
// Launch
// ---------------------------------------------------------------------------
extern "C" void kernel_launch(void* const* d_in, const int* in_sizes, int n_in,
                              void* d_out, int out_size, void* d_ws, size_t ws_size,
                              hipStream_t stream) {
    const float* x  = (const float*)d_in[0];
    const float* Wa = (const float*)d_in[1];
    const float* ba = (const float*)d_in[2];
    const float* Wp = (const float*)d_in[3];
    const float* bp = (const float*)d_in[4];
    float* out = (float*)d_out;

    unsigned short* ws = (unsigned short*)d_ws;
    // workspace layout (ushort elements), all 16B-aligned:
    unsigned short* xb  = ws;                         // 4096*1024
    unsigned short* wab = ws + 4194304;               // 1024*3072
    unsigned short* wpb = ws + 7340032;               // 1024*1024
    unsigned short* qb  = ws + 8388608;               // [B,H,T,HS]
    unsigned short* kb  = ws + 12582912;
    unsigned short* vb  = ws + 16777216;
    unsigned short* yb  = ws + 20971520;              // [B,T,C]

    cast_f32_bf16_kernel<<<(4194304 + 255) / 256, 256, 0, stream>>>(x, xb, 4194304);
    cast_f32_bf16_kernel<<<(3145728 + 255) / 256, 256, 0, stream>>>(Wa, wab, 3145728);
    cast_f32_bf16_kernel<<<(1048576 + 255) / 256, 256, 0, stream>>>(Wp, wpb, 1048576);

    qkv_gemm_kernel<<<768, 256, 0, stream>>>(xb, wab, ba, qb, kb, vb);
    flash_attn_kernel<<<1024, 128, 0, stream>>>(qb, kb, vb, yb);
    proj_gemm_kernel<<<256, 256, 0, stream>>>(yb, wpb, bp, out);
}
